// Hamburger_35811437314912
// MI455X (gfx1250) — matrix operand order repaired and checked by reading, not compiled, not run on hardware
//
#include <hip/hip_runtime.h>
#include <hip/hip_bf16.h>

// ---------------------------------------------------------------------------
// Hamburger (HamNet) block for MI455X / gfx1250, wave32.
//  - All GEMM math via v_wmma_f32_16x16x32_bf16 (bf16 operands, f32 accum).
//  - Every GEMM is NT: A (MxK) row-major, BT (NxK) row-major.
//  - GEMM stages 64x32 A/B tiles into LDS with global_load_async_to_lds_b128
//    (ASYNCcnt), double-buffered; fragments are read back as ds_load_b128.
//  - bf16 operand matrices (Y/YT ~75MB each) fit in the 192MB L2, so the six
//    NMF sweeps re-read from L2, not HBM.
// ---------------------------------------------------------------------------

typedef __bf16 bf16_t;
typedef __attribute__((ext_vector_type(16))) __bf16 v16bf;
typedef __attribute__((ext_vector_type(8)))  __bf16 v8bf;
typedef __attribute__((ext_vector_type(8)))  float  v8f;

// gfx1250 async global->LDS copy (GV form, 16B per lane). VDST = LDS byte
// address (wave-relative; low 32 bits of the flat pointer), VADDR = 64-bit
// global address. Tracked by ASYNCcnt.
static __device__ __forceinline__ void async_copy_b128(unsigned lds_addr,
                                                       const void* gptr)
{
  asm volatile("global_load_async_to_lds_b128 %0, %1, off"
               :: "v"(lds_addr), "v"((unsigned long long)(uintptr_t)gptr)
               : "memory");
}

static __device__ __forceinline__ void wait_asynccnt0()
{
  asm volatile("s_wait_asynccnt 0" ::: "memory");
}

// ---------------------------------------------------------------------------
// NT GEMM:  C[m,n] = sum_k A[m,k] * BT[n,k]  (+bias[n]), batched over blockIdx.z
// Block: 128 threads = 4 waves; block tile 64x64; wave tile 32x32 (2x2 WMMA).
// Requires: M % 64 == 0, Nc % 64 == 0, K % 32 == 0 (true for all call sites).
// ---------------------------------------------------------------------------
#define TILE_K 32
#define TILE_BYTES (64 * TILE_K * 2)   // 4096B per matrix tile

__global__ __launch_bounds__(128)
void gemm_nt_wmma(const bf16_t* __restrict__ A, const bf16_t* __restrict__ BT,
                  float* __restrict__ Cf, bf16_t* __restrict__ Cb,
                  const float* __restrict__ bias,
                  int M, int Nc, int K,
                  long long sA, long long sB, long long sC)
{
  __shared__ bf16_t sAt[2][64 * TILE_K];   // [buf][row*32 + k]
  __shared__ bf16_t sBt[2][64 * TILE_K];

  const int bz = blockIdx.z;
  A  += (size_t)bz * sA;
  BT += (size_t)bz * sB;

  const int lane = threadIdx.x & 31;
  const int wave = threadIdx.x >> 5;
  const int wm   = (wave >> 1) * 32;            // wave row origin within block
  const int wn   = (wave & 1) * 32;             // wave col origin within block
  const int bm0  = blockIdx.y * 64;             // block row origin
  const int bn0  = blockIdx.x * 64;             // block col origin
  const int r16  = lane & 15;
  const int half = lane >> 4;

  // Async-copy addressing: 128 threads x 32B cover one 4KB tile per matrix.
  const int t    = threadIdx.x;
  const int crow = t >> 1;                      // 0..63
  const int cseg = (t & 1) * 16;                // halves 0..15 / 16..31
  const bf16_t* gA = A  + (size_t)(bm0 + crow) * K + cseg;
  const bf16_t* gB = BT + (size_t)(bn0 + crow) * K + cseg;
  const unsigned ldsA0 = (unsigned)(uintptr_t)(&sAt[0][0]) +
                         (unsigned)(crow * TILE_K + cseg) * 2u;
  const unsigned ldsB0 = (unsigned)(uintptr_t)(&sBt[0][0]) +
                         (unsigned)(crow * TILE_K + cseg) * 2u;

  // Prologue: stage tile for kk = 0 into buffer 0.
  async_copy_b128(ldsA0,      gA);
  async_copy_b128(ldsA0 + 16, gA + 8);
  async_copy_b128(ldsB0,      gB);
  async_copy_b128(ldsB0 + 16, gB + 8);

  v8f acc[2][2] = {};

  int buf = 0;
  for (int kk = 0; kk < K; kk += TILE_K, buf ^= 1) {
    wait_asynccnt0();          // my slice of the current tile has landed
    __syncthreads();           // whole tile visible; prior reads of other buf done

    if (kk + TILE_K < K) {     // issue next tile into the other buffer
      const unsigned nb = (unsigned)((buf ^ 1) * TILE_BYTES);
      async_copy_b128(ldsA0 + nb,      gA + kk + TILE_K);
      async_copy_b128(ldsA0 + nb + 16, gA + kk + TILE_K + 8);
      async_copy_b128(ldsB0 + nb,      gB + kk + TILE_K);
      async_copy_b128(ldsB0 + nb + 16, gB + kk + TILE_K + 8);
    }

    const bf16_t* tA = &sAt[buf][0];
    const bf16_t* tB = &sBt[buf][0];
    v16bf af[2], bfr[2];
#pragma unroll
    for (int i = 0; i < 2; ++i) {
      // A fragment (16x32): elems 0..7 = K[half*8..+7], 8..15 = K[16+half*8..]
      const bf16_t* pa = tA + (wm + i * 16 + r16) * TILE_K + half * 8;
      v8bf alo = *(const v8bf*)(pa);
      v8bf ahi = *(const v8bf*)(pa + 16);
      // B fragment (32x16): elems 0..7 = K[half*8..+7], 8..15 = K[16+half*8..]
      const bf16_t* pb = tB + (wn + i * 16 + r16) * TILE_K + half * 8;
      v8bf blo = *(const v8bf*)(pb);
      v8bf bhi = *(const v8bf*)(pb + 16);
#pragma unroll
      for (int e = 0; e < 8; ++e) {
        af[i][e]      = alo[e];
        af[i][e + 8]  = ahi[e];
        bfr[i][e]     = blo[e];
        bfr[i][e + 8] = bhi[e];
      }
    }
#pragma unroll
    for (int im = 0; im < 2; ++im)
#pragma unroll
      for (int in = 0; in < 2; ++in)
        acc[im][in] = __builtin_amdgcn_wmma_f32_16x16x32_bf16(
            false, af[im], false, bfr[in], (short)0, acc[im][in], false, false);
  }

  // Epilogue: C/D layout -> col n = r16, row m = v + 8*half (per 16x16 tile)
#pragma unroll
  for (int im = 0; im < 2; ++im) {
#pragma unroll
    for (int in = 0; in < 2; ++in) {
      const int n = bn0 + wn + in * 16 + r16;
      const float badd = bias ? bias[n] : 0.0f;
#pragma unroll
      for (int v = 0; v < 8; ++v) {
        const int m = bm0 + wm + im * 16 + v + 8 * half;
        const float val = acc[im][in][v] + badd;
        const size_t off = (size_t)bz * sC + (size_t)m * Nc + n;
        if (Cf) Cf[off] = val;
        else    Cb[off] = (bf16_t)val;
      }
    }
  }
}

// ---------------------------------------------------------------------------
// Elementwise / utility kernels
// ---------------------------------------------------------------------------
__global__ void cast_bf16_kernel(const float* __restrict__ in,
                                 bf16_t* __restrict__ out, int n)
{
  for (int i = blockIdx.x * blockDim.x + threadIdx.x; i < n;
       i += gridDim.x * blockDim.x)
    out[i] = (bf16_t)in[i];
}

// in (rows x cols, f32) -> out (cols x rows, bf16), batched over z
__global__ __launch_bounds__(256)
void transpose_cast_kernel(const float* __restrict__ in, bf16_t* __restrict__ out,
                           int rows, int cols)
{
  __shared__ float t[32][33];
  const size_t base = (size_t)blockIdx.z * rows * cols;
  const int x0 = blockIdx.x * 32, y0 = blockIdx.y * 32;
  const int tx = threadIdx.x, ty = threadIdx.y;   // block (32,8)
#pragma unroll
  for (int j = 0; j < 4; ++j) {
    int r = y0 + ty + j * 8, c = x0 + tx;
    if (r < rows && c < cols) t[ty + j * 8][tx] = in[base + (size_t)r * cols + c];
  }
  __syncthreads();
#pragma unroll
  for (int j = 0; j < 4; ++j) {
    int orow = x0 + ty + j * 8, ocol = y0 + tx;   // out is cols x rows
    if (orow < cols && ocol < rows)
      out[base + (size_t)orow * rows + ocol] = (bf16_t)t[tx][ty + j * 8];
  }
}

// in (rows x cols, bf16) -> out (cols x rows, bf16), batched over z
__global__ __launch_bounds__(256)
void transpose_bf16_kernel(const bf16_t* __restrict__ in, bf16_t* __restrict__ out,
                           int rows, int cols)
{
  __shared__ bf16_t t[32][33];
  const size_t base = (size_t)blockIdx.z * rows * cols;
  const int x0 = blockIdx.x * 32, y0 = blockIdx.y * 32;
  const int tx = threadIdx.x, ty = threadIdx.y;
#pragma unroll
  for (int j = 0; j < 4; ++j) {
    int r = y0 + ty + j * 8, c = x0 + tx;
    if (r < rows && c < cols) t[ty + j * 8][tx] = in[base + (size_t)r * cols + c];
  }
  __syncthreads();
#pragma unroll
  for (int j = 0; j < 4; ++j) {
    int orow = x0 + ty + j * 8, ocol = y0 + tx;
    if (orow < cols && ocol < rows)
      out[base + (size_t)orow * rows + ocol] = t[tx][ty + j * 8];
  }
}

static __device__ __forceinline__ float hash_u01(unsigned v)
{
  v ^= v >> 17; v *= 0xed5ad4bbu;
  v ^= v >> 11; v *= 0xac4c1b51u;
  v ^= v >> 15; v *= 0x31848babu;
  v ^= v >> 14;
  return (float)(v >> 8) * (1.0f / 16777216.0f);
}

__global__ void init_bases_kernel(float* __restrict__ basesF, int n)
{
  for (int i = blockIdx.x * blockDim.x + threadIdx.x; i < n;
       i += gridDim.x * blockDim.x)
    basesF[i] = hash_u01((unsigned)i * 2654435761u + 0x9E3779B9u);
}

// One block per (b,r): L2-normalize bases column over D, emit f32 + both bf16 layouts
__global__ __launch_bounds__(256)
void normalize_bases_kernel(float* __restrict__ basesF,
                            bf16_t* __restrict__ bb, bf16_t* __restrict__ bt,
                            int D, int R)
{
  const int b = blockIdx.x / R, r = blockIdx.x % R, d = threadIdx.x;
  const size_t base = (size_t)b * D * R;
  float v = basesF[base + (size_t)d * R + r];
  __shared__ float red[256];
  red[d] = v * v;
  __syncthreads();
  for (int s = 128; s > 0; s >>= 1) {
    if (d < s) red[d] += red[d + s];
    __syncthreads();
  }
  const float norm = fmaxf(sqrtf(red[0]), 1e-12f);
  v /= norm;
  basesF[base + (size_t)d * R + r] = v;
  bb[base + (size_t)d * R + r]     = (bf16_t)v;
  bt[(size_t)b * R * D + (size_t)r * D + d] = (bf16_t)v;
}

// softmax over R (=64) per row of num (B*N rows); writes coefF, coef_bf, coefT_bf
__global__ __launch_bounds__(64)
void softmax_rows_kernel(const float* __restrict__ num, float* __restrict__ coefF,
                         bf16_t* __restrict__ cb, bf16_t* __restrict__ ct,
                         int Nn, int R)
{
  const int row = blockIdx.x;        // b*Nn + n
  const int r = threadIdx.x;         // 0..63
  float v = num[(size_t)row * R + r];
  __shared__ float red[64];
  red[r] = v; __syncthreads();
  for (int s = 32; s > 0; s >>= 1) {
    if (r < s) red[r] = fmaxf(red[r], red[r + s]);
    __syncthreads();
  }
  const float mx = red[0];
  __syncthreads();
  const float e = __expf(v - mx);
  red[r] = e; __syncthreads();
  for (int s = 32; s > 0; s >>= 1) {
    if (r < s) red[r] += red[r + s];
    __syncthreads();
  }
  const float c = e / red[0];
  coefF[(size_t)row * R + r] = c;
  cb[(size_t)row * R + r]    = (bf16_t)c;
  const int b = row / Nn, n = row - b * Nn;
  ct[((size_t)b * R + r) * Nn + n] = (bf16_t)c;
}

// coef *= num / (den + eps); refresh bf16 (N,R) and transposed bf16 (R,N)
__global__ void coef_update_kernel(float* __restrict__ coefF,
                                   const float* __restrict__ num,
                                   const float* __restrict__ den,
                                   bf16_t* __restrict__ cb, bf16_t* __restrict__ ct,
                                   int Nn, int R, int total)
{
  for (int idx = blockIdx.x * blockDim.x + threadIdx.x; idx < total;
       idx += gridDim.x * blockDim.x) {
    const float v = coefF[idx] * num[idx] / (den[idx] + 1e-6f);
    coefF[idx] = v;
    cb[idx] = (bf16_t)v;
    const int b = idx / (Nn * R);
    const int rem = idx - b * Nn * R;
    const int n = rem / R, r = rem - n * R;
    ct[((size_t)b * R + r) * Nn + n] = (bf16_t)v;
  }
}

// bases *= num / (den + eps); refresh bf16 (D,R) and transposed bf16 (R,D)
__global__ void bases_update_kernel(float* __restrict__ basesF,
                                    const float* __restrict__ num,
                                    const float* __restrict__ den,
                                    bf16_t* __restrict__ bb, bf16_t* __restrict__ bt,
                                    int D, int R, int total)
{
  for (int idx = blockIdx.x * blockDim.x + threadIdx.x; idx < total;
       idx += gridDim.x * blockDim.x) {
    const float v = basesF[idx] * num[idx] / (den[idx] + 1e-6f);
    basesF[idx] = v;
    bb[idx] = (bf16_t)v;
    const int b = idx / (D * R);
    const int rem = idx - b * D * R;
    const int d = rem / R, r = rem - d * R;
    bt[((size_t)b * R + r) * D + d] = (bf16_t)v;
  }
}

// GroupNorm stats: one block per (b,g); ZT layout is (b, n, c)
__global__ __launch_bounds__(256)
void gn_reduce_kernel(const float* __restrict__ ZT, float* __restrict__ mean,
                      float* __restrict__ var, int Nn, int C, int chPerG)
{
  const int bg = blockIdx.x;
  const int b = bg >> 5, g = bg & 31;
  const size_t base = (size_t)b * Nn * C + (size_t)g * chPerG;
  const int cnt = Nn * chPerG;
  float s = 0.f, ss = 0.f;
  for (int i = threadIdx.x; i < cnt; i += 256) {
    const int n = i / chPerG, c = i - n * chPerG;
    const float v = ZT[base + (size_t)n * C + c];
    s += v; ss += v * v;
  }
  __shared__ float rs[256], rss[256];
  rs[threadIdx.x] = s; rss[threadIdx.x] = ss;
  __syncthreads();
  for (int st = 128; st > 0; st >>= 1) {
    if (threadIdx.x < st) {
      rs[threadIdx.x] += rs[threadIdx.x + st];
      rss[threadIdx.x] += rss[threadIdx.x + st];
    }
    __syncthreads();
  }
  if (threadIdx.x == 0) {
    const float mu = rs[0] / cnt;
    mean[bg] = mu;
    var[bg] = rss[0] / cnt - mu * mu;
  }
}

// out[b,c,n] = relu(x[b,c,n] + gn(ZT)[b,n,c])
__global__ void gn_apply_kernel(const float* __restrict__ x,
                                const float* __restrict__ ZT,
                                const float* __restrict__ mean,
                                const float* __restrict__ var,
                                const float* __restrict__ gamma,
                                const float* __restrict__ beta,
                                float* __restrict__ out, int C, int Nn, int total)
{
  for (int idx = blockIdx.x * blockDim.x + threadIdx.x; idx < total;
       idx += gridDim.x * blockDim.x) {
    const int b = idx / (C * Nn);
    const int rem = idx - b * C * Nn;
    const int c = rem / Nn, n = rem - c * Nn;
    const int g = c >> 3;                         // C/32 = 8 channels per group
    const float mu = mean[b * 32 + g];
    const float vv = var[b * 32 + g];
    const float z = ZT[((size_t)b * Nn + n) * C + c];
    const float y = (z - mu) * rsqrtf(vv + 1e-5f) * gamma[c] + beta[c];
    out[idx] = fmaxf(x[idx] + y, 0.0f);
  }
}

// ---------------------------------------------------------------------------
// Host orchestration
// ---------------------------------------------------------------------------
static void launch_gemm(const bf16_t* A, const bf16_t* BT, float* Cf, bf16_t* Cb,
                        const float* bias, int M, int Nc, int K,
                        long long sA, long long sB, long long sC, int batch,
                        hipStream_t s)
{
  dim3 g(Nc / 64, M / 64, batch), b(128);
  gemm_nt_wmma<<<g, b, 0, s>>>(A, BT, Cf, Cb, bias, M, Nc, K, sA, sB, sC);
}

extern "C" void kernel_launch(void* const* d_in, const int* in_sizes, int n_in,
                              void* d_out, int out_size, void* d_ws, size_t ws_size,
                              hipStream_t stream)
{
  const float* x     = (const float*)d_in[0];   // (B,C,H,W)
  const float* w_in  = (const float*)d_in[1];   // (C,C)
  const float* b_in  = (const float*)d_in[2];   // (C)
  const float* w_out = (const float*)d_in[3];   // (C,C)
  const float* gamma = (const float*)d_in[4];   // (C)
  const float* beta  = (const float*)d_in[5];   // (C)
  float* out = (float*)d_out;

  const int B = 16, C = 256, Nn = 96 * 96, D = 256, R = 64;
  const size_t elW     = (size_t)C * C;          // 65536
  const size_t elBig   = (size_t)B * Nn * C;     // 37.7M (per bf16 buf: 75.5MB)
  const size_t elBases = (size_t)B * D * R;      // 262144
  const size_t elCoef  = (size_t)B * Nn * R;     // 9.4M
  const size_t elRR    = (size_t)B * R * R;      // 65536

  unsigned char* p = (unsigned char*)d_ws;
  auto carve = [&](size_t bytes) -> void* {
    void* r = (void*)p;
    p += (bytes + 255) & ~(size_t)255;
    return r;
  };
  bf16_t* w_in_bf  = (bf16_t*)carve(elW * 2);
  bf16_t* w_out_bf = (bf16_t*)carve(elW * 2);
  bf16_t* XT       = (bf16_t*)carve(elBig * 2);  // x^T per batch; reused as recon^T
  bf16_t* YT       = (bf16_t*)carve(elBig * 2);  // (N,C) bf16
  bf16_t* Yb       = (bf16_t*)carve(elBig * 2);  // (C,N) bf16
  float*  basesF   = (float*)carve(elBases * 4);
  bf16_t* bases_bf = (bf16_t*)carve(elBases * 2);  // (D,R)
  bf16_t* basesT_bf= (bf16_t*)carve(elBases * 2);  // (R,D)
  float*  coefF    = (float*)carve(elCoef * 4);
  bf16_t* coef_bf  = (bf16_t*)carve(elCoef * 2);   // (N,R)
  bf16_t* coefT_bf = (bf16_t*)carve(elCoef * 2);   // (R,N)
  float*  numC     = (float*)carve(elCoef * 4);
  float*  denC     = (float*)carve(elCoef * 4);
  float*  numB     = (float*)carve(elBases * 4);
  float*  denB     = (float*)carve(elBases * 4);
  bf16_t* brs      = (bf16_t*)carve(elRR * 2);     // bases^T bases (symmetric)
  bf16_t* crr      = (bf16_t*)carve(elRR * 2);     // coef^T coef (symmetric)
  float*  ZT       = (float*)carve(elBig * 4);     // (B,N,C) f32
  float*  gmean    = (float*)carve((size_t)B * 32 * 4);
  float*  gvar     = (float*)carve((size_t)B * 32 * 4);
  (void)ws_size; (void)in_sizes; (void)n_in; (void)out_size;

  const long long sBig = (long long)Nn * C;
  const long long sBas = (long long)D * R;
  const long long sCo  = (long long)Nn * R;
  const long long sRR  = (long long)R * R;

  // 1) weights -> bf16
  cast_bf16_kernel<<<256, 256, 0, stream>>>(w_in, w_in_bf, (int)elW);
  cast_bf16_kernel<<<256, 256, 0, stream>>>(w_out, w_out_bf, (int)elW);

  // 2) x (C,N) -> XT (N,C) bf16, per batch
  {
    dim3 g(Nn / 32, C / 32, B), b(32, 8);
    transpose_cast_kernel<<<g, b, 0, stream>>>(x, XT, C, Nn);
  }

  // 3) ham_in (as N-major): YT[n,o] = sum_c XT[n,c]*w_in[o,c] + b_in[o]
  launch_gemm(XT, w_in_bf, nullptr, YT, b_in, Nn, C, C, sBig, 0, sBig, B, stream);

  // 4) Yb (C,N) = transpose(YT)
  {
    dim3 g(C / 32, Nn / 32, B), b(32, 8);
    transpose_bf16_kernel<<<g, b, 0, stream>>>(YT, Yb, Nn, C);
  }

  // 5) bases init (uniform hash RNG) + column L2 normalize
  init_bases_kernel<<<1024, 256, 0, stream>>>(basesF, (int)elBases);
  normalize_bases_kernel<<<B * R, 256, 0, stream>>>(basesF, bases_bf, basesT_bf, D, R);

  // 6) coef init: numC = YT @ bases  -> softmax over R
  launch_gemm(YT, basesT_bf, numC, nullptr, nullptr, Nn, R, C, sBig, sBas, sCo, B, stream);
  softmax_rows_kernel<<<B * Nn, 64, 0, stream>>>(numC, coefF, coef_bf, coefT_bf, Nn, R);

  // 7) six multiplicative-update steps
  for (int step = 0; step < 6; ++step) {
    // coef update
    launch_gemm(YT, basesT_bf, numC, nullptr, nullptr, Nn, R, C, sBig, sBas, sCo, B, stream);
    launch_gemm(basesT_bf, basesT_bf, nullptr, brs, nullptr, R, R, C, sBas, sBas, sRR, B, stream);
    launch_gemm(coef_bf, brs, denC, nullptr, nullptr, Nn, R, R, sCo, sRR, sCo, B, stream);
    coef_update_kernel<<<(int)((elCoef + 255) / 256), 256, 0, stream>>>(
        coefF, numC, denC, coef_bf, coefT_bf, Nn, R, (int)elCoef);
    // bases update
    launch_gemm(Yb, coefT_bf, numB, nullptr, nullptr, D, R, Nn, sBig, sCo, sBas, B, stream);
    launch_gemm(coefT_bf, coefT_bf, nullptr, crr, nullptr, R, R, Nn, sCo, sCo, sRR, B, stream);
    launch_gemm(bases_bf, crr, denB, nullptr, nullptr, D, R, R, sBas, sRR, sBas, B, stream);
    bases_update_kernel<<<1024, 256, 0, stream>>>(
        basesF, numB, denB, bases_bf, basesT_bf, D, R, (int)elBases);
  }

  // 8) final coef refresh (reference's post-scan compute_coef)
  launch_gemm(YT, basesT_bf, numC, nullptr, nullptr, Nn, R, C, sBig, sBas, sCo, B, stream);
  launch_gemm(basesT_bf, basesT_bf, nullptr, brs, nullptr, R, R, C, sBas, sBas, sRR, B, stream);
  launch_gemm(coef_bf, brs, denC, nullptr, nullptr, Nn, R, R, sCo, sRR, sCo, B, stream);
  coef_update_kernel<<<(int)((elCoef + 255) / 256), 256, 0, stream>>>(
      coefF, numC, denC, coef_bf, coefT_bf, Nn, R, (int)elCoef);

  // 9) recon^T (N,C) = coef (N,R) @ bases^T  -> reuse XT buffer (bf16)
  launch_gemm(coef_bf, bases_bf, nullptr, XT, nullptr, Nn, C, R, sCo, sBas, sBig, B, stream);

  // 10) ham_out (N-major): ZT[n,o] = sum_c reconT[n,c]*w_out[o,c]
  launch_gemm(XT, w_out_bf, ZT, nullptr, nullptr, Nn, C, C, sBig, 0, sBig, B, stream);

  // 11) GroupNorm stats + apply + residual + relu
  gn_reduce_kernel<<<B * 32, 256, 0, stream>>>(ZT, gmean, gvar, Nn, C, C / 32);
  const int total = (int)((size_t)B * C * Nn);
  gn_apply_kernel<<<(total + 255) / 256, 256, 0, stream>>>(
      x, ZT, gmean, gvar, gamma, beta, out, C, Nn, total);
}
